// MixSample_76020921139873
// MI455X (gfx1250) — matrix operand order repaired and checked by reading, not compiled, not run on hardware
//
#include <hip/hip_runtime.h>

typedef __attribute__((ext_vector_type(16))) _Float16 v16h;
typedef __attribute__((ext_vector_type(8)))  float    v8f;

#define N_      16
#define C_      64
#define T_      30
#define H_      64
#define W_      44
#define OUT_H_  64
#define OUT_W_  44

// workspace layout, in float (32-bit) units
#define FY_OFF  0                         // [16][64][64] f32
#define FX_OFF  (16 * 64 * 64)            // [16][44][44] f32
#define I0_OFF  (FX_OFF + 16 * 44 * 44)   // [16][30] int
#define I1_OFF  (I0_OFF + 16 * 30)        // [16][30] int
#define A0_OFF  (I1_OFF + 16 * 30)        // [16][30] f32
#define A1_OFF  (A0_OFF + 16 * 30)        // [16][30] f32

// ---------------------------------------------------------------------------
// Prologue 1: temporal resampling coefficients per (n, to).
// a0/a1 fold in the 0.5 bilinear factor, the boundary masks, and gamma.
// ---------------------------------------------------------------------------
__global__ void mix_params_kernel(const float* __restrict__ dt,
                                  const float* __restrict__ log_delta_t,
                                  const float* __restrict__ log_gamma,
                                  float* __restrict__ ws) {
    int idx = threadIdx.x;
    if (idx >= N_ * T_) return;
    int n  = idx / T_;
    int to = idx % T_;

    float dt1     = tanhf(dt[n]) * (T_ * 0.5f) + T_ * 0.5f;
    float delta_t = expf(log_delta_t[n]);
    float gamma   = 1.0f / (1.0f + expf(-log_gamma[n]));
    float dt2     = tanhf(dt1) * (T_ * 0.5f) + 15.0f;        // anchor_t = round(T/2) = 15

    float mu_t  = dt2 + ((float)to - T_ * 0.5f) * delta_t;
    float coord = mu_t / (float)(T_ - 1) * 2.0f - 1.0f;
    float iy    = ((coord + 1.0f) * (float)T_ - 1.0f) * 0.5f;

    float i0f = floorf(iy);
    float w1  = iy - i0f;
    float w0  = 1.0f - w1;
    int   i0  = (int)i0f;
    int   i1  = i0 + 1;
    float m0  = (i0 >= 0 && i0 < T_) ? 1.0f : 0.0f;
    float m1  = (i1 >= 0 && i1 < T_) ? 1.0f : 0.0f;
    int   i0c = min(max(i0, 0), T_ - 1);
    int   i1c = min(max(i1, 0), T_ - 1);

    ((int*)ws)[I0_OFF + idx] = i0c;
    ((int*)ws)[I1_OFF + idx] = i1c;
    ws[A0_OFF + idx] = 0.5f * gamma * w0 * m0;
    ws[A1_OFF + idx] = 0.5f * gamma * w1 * m1;
}

// ---------------------------------------------------------------------------
// Prologue 2: normalized Gaussian filterbanks Fy [n][64][64], Fx [n][44][44].
// One block per sample n; one thread per filter row (108 rows).
// ---------------------------------------------------------------------------
__global__ void filterbank_kernel(const float* __restrict__ dx,
                                  const float* __restrict__ dy,
                                  const float* __restrict__ log_sigma2,
                                  const float* __restrict__ log_delta,
                                  float* __restrict__ ws) {
    int n = blockIdx.x;
    int r = threadIdx.x;
    float sigma2 = expf(log_sigma2[n]);
    float delta  = expf(log_delta[n]);
    float inv2s  = 1.0f / (2.0f * sigma2);

    if (r < OUT_H_) {                    // Fy row p = r, over h = 0..63
        float mu = tanhf(dy[n]) * 32.0f + 32.0f + ((float)r - 32.0f) * delta;
        float s = 0.0f;
        for (int h = 0; h < H_; ++h) { float d = (float)h - mu; s += expf(-d * d * inv2s); }
        float inv = 1.0f / fmaxf(s, 1e-8f);
        for (int h = 0; h < H_; ++h) {
            float d = (float)h - mu;
            ws[FY_OFF + (n * OUT_H_ + r) * H_ + h] = expf(-d * d * inv2s) * inv;
        }
    } else if (r < OUT_H_ + OUT_W_) {    // Fx row q, over w = 0..43
        int q = r - OUT_H_;
        float mu = tanhf(dx[n]) * 22.0f + 22.0f + ((float)q - 22.0f) * delta;
        float s = 0.0f;
        for (int w = 0; w < W_; ++w) { float d = (float)w - mu; s += expf(-d * d * inv2s); }
        float inv = 1.0f / fmaxf(s, 1e-8f);
        for (int w = 0; w < W_; ++w) {
            float d = (float)w - mu;
            ws[FX_OFF + (n * OUT_W_ + q) * W_ + w] = expf(-d * d * inv2s) * inv;
        }
    }
}

// ---------------------------------------------------------------------------
// Main fused kernel: one block per (n, c, to).
//   Xmix = a0 * x[n,c,i0] + a1 * x[n,c,i1]           (64x44 -> f16 LDS, padded)
//   tmp  = Xmix * Fx^T   (M=64 h, N=48 q, K=64 w)    via v_wmma_f32_16x16x32_f16
//   out  = Fy * tmp      (M=64 p, N=48 q, K=64 h)    via v_wmma_f32_16x16x32_f16
// 128 threads = 4 waves; wave w owns M strip [16w,16w+16), 3 N tiles, 2 K steps.
// ---------------------------------------------------------------------------
#define ST 66   // LDS row stride in halves: 33-dword rows avoid bank conflicts

__global__ __launch_bounds__(128) void glimpse_kernel(const float* __restrict__ x,
                                                      const float* __restrict__ ws,
                                                      float* __restrict__ out) {
    __shared__ _Float16 Xs[64 * ST];    // [h][w]  (A, stage 1)
    __shared__ _Float16 Fys[64 * ST];   // [p][h]  (A, stage 2)
    __shared__ _Float16 FxQ[48 * ST];   // [q][w]  (B, stage 1) = Fx^T by layout
    __shared__ _Float16 TmpT[48 * ST];  // [q][h]  (B, stage 2)

    int b   = blockIdx.x;
    int to  = b % T_;
    int c   = (b / T_) % C_;
    int n   = b / (T_ * C_);
    int tid  = threadIdx.x;
    int lane = tid & 31;
    int wv   = tid >> 5;        // wave id 0..3
    int hlf  = lane >> 4;       // lane half (K split)
    int lidx = lane & 15;       // M (A) / N (B,D) index within tile

    const int* iws = (const int*)ws;
    int   pidx = n * T_ + to;
    int   i0   = iws[I0_OFF + pidx];
    int   i1   = iws[I1_OFF + pidx];
    float a0   = ws[A0_OFF + pidx];
    float a1   = ws[A1_OFF + pidx];

    const float* x0 = x + (((size_t)(n * C_ + c)) * T_ + i0) * (size_t)(H_ * W_);
    const float* x1 = x + (((size_t)(n * C_ + c)) * T_ + i1) * (size_t)(H_ * W_);

    // blend two temporal slices -> f16, zero-pad w to 64
    for (int i = tid; i < 64 * 64; i += 128) {
        int h = i >> 6, w = i & 63;
        float v = (w < W_) ? (a0 * x0[h * W_ + w] + a1 * x1[h * W_ + w]) : 0.0f;
        Xs[h * ST + w] = (_Float16)v;
    }
    const float* Fy = ws + FY_OFF + n * 64 * 64;
    for (int i = tid; i < 64 * 64; i += 128) {
        int p = i >> 6, h = i & 63;
        Fys[p * ST + h] = (_Float16)Fy[p * 64 + h];
    }
    const float* Fx = ws + FX_OFF + n * 44 * 44;
    for (int i = tid; i < 48 * 64; i += 128) {
        int q = i >> 6, w = i & 63;
        float v = (q < W_ && w < W_) ? Fx[q * 44 + w] : 0.0f;
        FxQ[q * ST + w] = (_Float16)v;
    }
    __syncthreads();

    // -------- stage 1: tmp[h][q] = sum_w Xs[h][w] * FxQ[q][w] --------
    {
        v8f acc0 = {}; v8f acc1 = {}; v8f acc2 = {};
        int m0 = wv * 16;
        #pragma unroll
        for (int ks = 0; ks < 2; ++ks) {
            union { v16h v; unsigned int u[8]; } A;
            int rowA = (m0 + lidx) * ST;
            #pragma unroll
            for (int vv = 0; vv < 8; ++vv) {
                int k = (vv & 3) * 2 + (vv >> 2) * 16 + hlf * 8 + ks * 32;
                A.u[vv] = *(const unsigned int*)&Xs[rowA + k];
            }
            #pragma unroll
            for (int nt = 0; nt < 3; ++nt) {
                union { v16h v; unsigned int u[8]; } B;
                int rowB = (nt * 16 + lidx) * ST;
                #pragma unroll
                for (int vv = 0; vv < 8; ++vv) {
                    int k = (vv & 3) * 2 + (vv >> 2) * 16 + hlf * 8 + ks * 32;
                    B.u[vv] = *(const unsigned int*)&FxQ[rowB + k];
                }
                v8f cin = (nt == 0) ? acc0 : ((nt == 1) ? acc1 : acc2);
                v8f d = __builtin_amdgcn_wmma_f32_16x16x32_f16(
                            false, A.v, false, B.v, (short)0, cin, false, false);
                if (nt == 0) acc0 = d; else if (nt == 1) acc1 = d; else acc2 = d;
            }
        }
        // D layout: lane -> (m = r + 8*hlf, n = lidx); store transposed as TmpT[q][h]
        #pragma unroll
        for (int r = 0; r < 8; ++r) {
            int h = m0 + r + hlf * 8;
            TmpT[(0 * 16 + lidx) * ST + h] = (_Float16)acc0[r];
            TmpT[(1 * 16 + lidx) * ST + h] = (_Float16)acc1[r];
            TmpT[(2 * 16 + lidx) * ST + h] = (_Float16)acc2[r];
        }
    }
    __syncthreads();

    // -------- stage 2: G[p][q] = sum_h Fys[p][h] * TmpT[q][h] --------
    {
        v8f acc0 = {}; v8f acc1 = {}; v8f acc2 = {};
        int p0 = wv * 16;
        #pragma unroll
        for (int ks = 0; ks < 2; ++ks) {
            union { v16h v; unsigned int u[8]; } A;
            int rowA = (p0 + lidx) * ST;
            #pragma unroll
            for (int vv = 0; vv < 8; ++vv) {
                int k = (vv & 3) * 2 + (vv >> 2) * 16 + hlf * 8 + ks * 32;
                A.u[vv] = *(const unsigned int*)&Fys[rowA + k];
            }
            #pragma unroll
            for (int nt = 0; nt < 3; ++nt) {
                union { v16h v; unsigned int u[8]; } B;
                int rowB = (nt * 16 + lidx) * ST;
                #pragma unroll
                for (int vv = 0; vv < 8; ++vv) {
                    int k = (vv & 3) * 2 + (vv >> 2) * 16 + hlf * 8 + ks * 32;
                    B.u[vv] = *(const unsigned int*)&TmpT[rowB + k];
                }
                v8f cin = (nt == 0) ? acc0 : ((nt == 1) ? acc1 : acc2);
                v8f d = __builtin_amdgcn_wmma_f32_16x16x32_f16(
                            false, A.v, false, B.v, (short)0, cin, false, false);
                if (nt == 0) acc0 = d; else if (nt == 1) acc1 = d; else acc2 = d;
            }
        }
        // write out[n][c][to][p][q], q < 44 (temporal weights + gamma already folded in)
        float* outp = out + (((size_t)(n * C_ + c)) * T_ + to) * (size_t)(OUT_H_ * OUT_W_);
        #pragma unroll
        for (int nt = 0; nt < 3; ++nt) {
            int q = nt * 16 + lidx;
            if (q < OUT_W_) {
                #pragma unroll
                for (int r = 0; r < 8; ++r) {
                    int p = p0 + r + hlf * 8;
                    float v = (nt == 0) ? acc0[r] : ((nt == 1) ? acc1[r] : acc2[r]);
                    outp[p * OUT_W_ + q] = v;
                }
            }
        }
    }
}

// ---------------------------------------------------------------------------
extern "C" void kernel_launch(void* const* d_in, const int* in_sizes, int n_in,
                              void* d_out, int out_size, void* d_ws, size_t ws_size,
                              hipStream_t stream) {
    const float* x           = (const float*)d_in[0];
    const float* dt          = (const float*)d_in[1];
    const float* dx          = (const float*)d_in[2];
    const float* dy          = (const float*)d_in[3];
    const float* log_sigma2  = (const float*)d_in[4];
    const float* log_delta_t = (const float*)d_in[5];
    const float* log_delta   = (const float*)d_in[6];
    const float* log_gamma   = (const float*)d_in[7];
    float* ws  = (float*)d_ws;
    float* out = (float*)d_out;

    hipLaunchKernelGGL(mix_params_kernel, dim3(1), dim3(512), 0, stream,
                       dt, log_delta_t, log_gamma, ws);
    hipLaunchKernelGGL(filterbank_kernel, dim3(N_), dim3(128), 0, stream,
                       dx, dy, log_sigma2, log_delta, ws);
    hipLaunchKernelGGL(glimpse_kernel, dim3(N_ * C_ * T_), dim3(128), 0, stream,
                       x, ws, out);
}
